// NeuralGraphMatcher_14800457302144
// MI455X (gfx1250) — matrix operand order repaired and checked by reading, not compiled, not run on hardware
//
#include <hip/hip_runtime.h>

typedef __attribute__((ext_vector_type(16))) _Float16 v16h;
typedef __attribute__((ext_vector_type(8)))  _Float16 v8h;
typedef __attribute__((ext_vector_type(8)))  float    v8f;

constexpr int HDIM = 256;   // hidden size
constexpr int RDIM = 256;   // robots
constexpr int FDIM = 1024;  // frontiers

// ---------------------------------------------------------------------------
// Layer 1 (tiny K=4 or K=3): out[m][n] = relu(sum_k X[m][k]*W[k][n] + b[n]),
// stored as f16 row-major [M][H] so it can feed WMMA as the A matrix.
// ---------------------------------------------------------------------------
__global__ void layer1_kernel(const float* __restrict__ X,
                              const float* __restrict__ W,
                              const float* __restrict__ b,
                              _Float16* __restrict__ out,
                              int M, int Kin) {
  int idx = blockIdx.x * blockDim.x + threadIdx.x;
  if (idx >= M * HDIM) return;
  int m = idx / HDIM, n = idx % HDIM;
  float acc = b[n];
  for (int k = 0; k < Kin; ++k)
    acc = fmaf(X[m * Kin + k], W[k * HDIM + n], acc);
  out[idx] = (_Float16)fmaxf(acc, 0.0f);
}

// ---------------------------------------------------------------------------
// Weight prep: f32 [K][N] row-major -> f16 [N][K] ("Bt"), so each WMMA lane's
// B-fragment (one output column, two runs of 8 K, 16 apart) is two contiguous 16B runs.
// ---------------------------------------------------------------------------
__global__ void transpose_to_f16_kernel(const float* __restrict__ in,
                                        _Float16* __restrict__ out) {
  int idx = blockIdx.x * blockDim.x + threadIdx.x; // HDIM*HDIM threads
  int n = idx / HDIM;
  int k = idx % HDIM;
  out[n * HDIM + k] = (_Float16)in[k * HDIM + n];
}

// ---------------------------------------------------------------------------
// WMMA GEMM: C[M][N] = A[M][K](f16) @ B[K][N] (given as Bt[N][K] f16) + bias.
// One 16x16 tile per wave, K stepped by 32 via v_wmma_f32_16x16x32_f16.
// Fragment layouts per CDNA5 ISA 7.12.2 (wave32):
//   A: lane<16 row=lane, K elems {kk+e | e<8} and {kk+16+e-8 ...} -> two 8-half runs
//      offset by 8 for lanes 16..31.
//   B: lane = column (lane&15), same K order as A -> two 8-half runs.
//   C/D: elem e -> row m0+e+8*(lane>=16), col n0+(lane&15).
// ---------------------------------------------------------------------------
__global__ void wmma_gemm_kernel(const _Float16* __restrict__ A,   // [M][K]
                                 const _Float16* __restrict__ Bt,  // [N][K]
                                 const float* __restrict__ bias,   // [N] or null
                                 void* __restrict__ out,           // [M][N]
                                 int M, int N, int K, int out_half) {
  const int lane = threadIdx.x & 31;
  const int wave = threadIdx.x >> 5;
  const int rh   = lane >> 4;     // which 16-lane half
  const int l16  = lane & 15;
  const int m0 = blockIdx.y * 128 + wave * 16;
  const int n0 = blockIdx.x * 16;
  if (m0 >= M) return;

  const _Float16* Arow = A  + (size_t)(m0 + l16) * K;
  const _Float16* Bcol = Bt + (size_t)(n0 + l16) * K;

  v8f acc = {};
  for (int kk = 0; kk < K; kk += 32) {
    v8h a_lo = *(const v8h*)(Arow + kk + 8 * rh);
    v8h a_hi = *(const v8h*)(Arow + kk + 16 + 8 * rh);
    v8h b_lo = *(const v8h*)(Bcol + kk + 8 * rh);
    v8h b_hi = *(const v8h*)(Bcol + kk + 16 + 8 * rh);
    v16h a, b;
#pragma unroll
    for (int i = 0; i < 8; ++i) {
      a[i] = a_lo[i]; a[i + 8] = a_hi[i];
      b[i] = b_lo[i]; b[i + 8] = b_hi[i];
    }
    acc = __builtin_amdgcn_wmma_f32_16x16x32_f16(
        /*neg_a=*/false, a, /*neg_b=*/false, b,
        /*c_mod=*/(short)0, acc, /*reuse_a=*/false, /*reuse_b=*/false);
  }

  const float bv  = bias ? bias[n0 + l16] : 0.0f;
  const int   col = n0 + l16;
#pragma unroll
  for (int e = 0; e < 8; ++e) {
    int m = m0 + e + 8 * rh;
    float v = acc[e] + bv;
    if (out_half)
      ((_Float16*)out)[(size_t)m * N + col] = (_Float16)v;
    else
      ((float*)out)[(size_t)m * N + col] = v;
  }
}

// ---------------------------------------------------------------------------
// Fused final stage (avoids materializing the 268MB h tensor):
//   affinity[r,f] = sum_k relu(h_r[r,k] + h_f[f,k] + b1[k]) * w2[k] + b2
// One block per (robot r, 256-frontier slab). h_r row, b1, w2 cached in LDS;
// each thread streams one h_f row as float4 (L2-resident, reused R times).
// ---------------------------------------------------------------------------
__global__ void affinity_kernel(const float* __restrict__ h_r,   // [R][H]
                                const float* __restrict__ h_f,   // [F][H]
                                const float* __restrict__ a_b1,  // [H]
                                const float* __restrict__ a_w2,  // [H]
                                const float* __restrict__ a_b2,  // [1]
                                float* __restrict__ out) {       // [R][F]
  __shared__ float s_hr[HDIM], s_b[HDIM], s_w[HDIM];
  const int tid = threadIdx.x;
  const int r = blockIdx.y;
  s_hr[tid] = h_r[(size_t)r * HDIM + tid];
  s_b[tid]  = a_b1[tid];
  s_w[tid]  = a_w2[tid];
  __syncthreads();

  const int f = blockIdx.x * 256 + tid;
  const float4* hf = (const float4*)(h_f + (size_t)f * HDIM);
  float acc = 0.0f;
#pragma unroll 4
  for (int q = 0; q < HDIM / 4; ++q) {
    float4 v = hf[q];
    int k = q * 4;
    acc = fmaf(fmaxf(s_hr[k + 0] + v.x + s_b[k + 0], 0.0f), s_w[k + 0], acc);
    acc = fmaf(fmaxf(s_hr[k + 1] + v.y + s_b[k + 1], 0.0f), s_w[k + 1], acc);
    acc = fmaf(fmaxf(s_hr[k + 2] + v.z + s_b[k + 2], 0.0f), s_w[k + 2], acc);
    acc = fmaf(fmaxf(s_hr[k + 3] + v.w + s_b[k + 3], 0.0f), s_w[k + 3], acc);
  }
  out[(size_t)r * FDIM + f] = acc + a_b2[0];
}

// ---------------------------------------------------------------------------
extern "C" void kernel_launch(void* const* d_in, const int* in_sizes, int n_in,
                              void* d_out, int out_size, void* d_ws, size_t ws_size,
                              hipStream_t stream) {
  const float* robot_features    = (const float*)d_in[0];  // (256,4)
  const float* frontier_features = (const float*)d_in[1];  // (1024,3)
  const float* r_w1 = (const float*)d_in[2];               // (4,256)
  const float* r_b1 = (const float*)d_in[3];               // (256)
  const float* r_w2 = (const float*)d_in[4];               // (256,256)
  const float* r_b2 = (const float*)d_in[5];               // (256)
  const float* f_w1 = (const float*)d_in[6];               // (3,256)
  const float* f_b1 = (const float*)d_in[7];               // (256)
  const float* f_w2 = (const float*)d_in[8];               // (256,256)
  const float* f_b2 = (const float*)d_in[9];               // (256)
  const float* a_w1 = (const float*)d_in[10];              // (512,256)
  const float* a_b1 = (const float*)d_in[11];              // (256)
  const float* a_w2 = (const float*)d_in[12];              // (256,1)
  const float* a_b2 = (const float*)d_in[13];              // (1)
  float* out = (float*)d_out;                              // (256,1024)

  // Workspace layout (bytes)
  char* ws = (char*)d_ws;
  _Float16* A_r     = (_Float16*)(ws + 0);        // 256*256 f16   = 128KB
  _Float16* A_f     = (_Float16*)(ws + 131072);   // 1024*256 f16  = 512KB
  _Float16* w_r2t   = (_Float16*)(ws + 655360);   // 128KB
  _Float16* w_f2t   = (_Float16*)(ws + 786432);   // 128KB
  _Float16* w1rt    = (_Float16*)(ws + 917504);   // 128KB
  _Float16* w1ft    = (_Float16*)(ws + 1048576);  // 128KB
  _Float16* emb_r16 = (_Float16*)(ws + 1179648);  // 128KB
  _Float16* emb_f16 = (_Float16*)(ws + 1310720);  // 512KB
  float*    h_r     = (float*)   (ws + 1835008);  // 256KB
  float*    h_f     = (float*)   (ws + 2097152);  // 1MB   (total ~3MB)

  // 1) tiny first layers -> f16 activations
  layer1_kernel<<<(RDIM * HDIM) / 256, 256, 0, stream>>>(
      robot_features, r_w1, r_b1, A_r, RDIM, 4);
  layer1_kernel<<<(FDIM * HDIM) / 256, 256, 0, stream>>>(
      frontier_features, f_w1, f_b1, A_f, FDIM, 3);

  // 2) weight prep: transpose + f16 convert
  transpose_to_f16_kernel<<<(HDIM * HDIM) / 256, 256, 0, stream>>>(r_w2, w_r2t);
  transpose_to_f16_kernel<<<(HDIM * HDIM) / 256, 256, 0, stream>>>(f_w2, w_f2t);
  transpose_to_f16_kernel<<<(HDIM * HDIM) / 256, 256, 0, stream>>>(a_w1, w1rt);
  transpose_to_f16_kernel<<<(HDIM * HDIM) / 256, 256, 0, stream>>>(
      a_w1 + HDIM * HDIM, w1ft);

  // 3) WMMA GEMMs (16 col-tiles, 128 rows per block of 8 waves)
  dim3 blk(256);
  wmma_gemm_kernel<<<dim3(HDIM / 16, RDIM / 128), blk, 0, stream>>>(
      A_r, w_r2t, r_b2, emb_r16, RDIM, HDIM, HDIM, /*out_half=*/1);
  wmma_gemm_kernel<<<dim3(HDIM / 16, FDIM / 128), blk, 0, stream>>>(
      A_f, w_f2t, f_b2, emb_f16, FDIM, HDIM, HDIM, /*out_half=*/1);
  wmma_gemm_kernel<<<dim3(HDIM / 16, RDIM / 128), blk, 0, stream>>>(
      emb_r16, w1rt, nullptr, h_r, RDIM, HDIM, HDIM, /*out_half=*/0);
  wmma_gemm_kernel<<<dim3(HDIM / 16, FDIM / 128), blk, 0, stream>>>(
      emb_f16, w1ft, nullptr, h_f, FDIM, HDIM, HDIM, /*out_half=*/0);

  // 4) fused broadcast + relu + reduction epilogue
  affinity_kernel<<<dim3(FDIM / 256, RDIM), 256, 0, stream>>>(
      h_r, h_f, a_b1, a_w2, a_b2, out);
}
